// DifferentialAttention_18356690223144
// MI455X (gfx1250) — matrix operand order repaired and checked by reading, not compiled, not run on hardware
//
#include <hip/hip_runtime.h>

typedef _Float16 v4h  __attribute__((ext_vector_type(4)));
typedef _Float16 v8h  __attribute__((ext_vector_type(8)));
typedef _Float16 v16h __attribute__((ext_vector_type(16)));
typedef float    v8f  __attribute__((ext_vector_type(8)));
typedef __fp16   h2   __attribute__((ext_vector_type(2)));   // cvt_pkrtz result type

#define KSTR 72   // f16 units per K-tile row (64 + pad, keeps 16B alignment)
#define VSTR 72   // f16 units per Vt row   (multiple of 8 -> 8B/16B aligned)

union U16 { v16h v; v8h h[2]; };          // assemble WMMA operand from 2 LDS loads
union P16 { v16h v; h2 p[8]; };           // assemble WMMA operand from pkrtz pairs
union P8  { v8h  v; h2 p[4]; };
union P4  { v4h  v; h2 p[2]; };

__launch_bounds__(256, 2)
__global__ void diffattn_wmma(const float* __restrict__ Q,
                              const float* __restrict__ K,
                              const float* __restrict__ V,
                              const float* __restrict__ lambda_p,
                              float* __restrict__ Out)
{
    constexpr int   S = 2048, D = 64;
    // 1/sqrt(32) * log2(e): exp2-domain softmax (v_exp_f32 is base-2)
    constexpr float SCALE_LOG2 = 0.17677669529663688f * 1.4426950408889634f;

    __shared__ _Float16 Kl[64 * KSTR];   // K tile, row-major [key][d], f16
    __shared__ _Float16 Vt[64 * VSTR];   // V tile, transposed [d][key], f16

    const int tid  = threadIdx.x;
    const int lane = tid & 31;
    const int wave = tid >> 5;
    const int ln16 = lane & 15;
    const int hh   = lane >> 4;          // lane half (0/1)
    const int bh   = blockIdx.y;         // b*16 + h
    const float lam = lambda_p[bh & 15];

    const size_t base = (size_t)bh * S * D;
    const int q0 = blockIdx.x * 128 + wave * 16;   // 8 waves x 16 q-rows

    // ---- Q as WMMA B-operand (32 x 16: K=d, N=q), per half j. One row/lane,
    // contiguous 16 f16 at d = 32j + 16*hh. Scale folded into conversion. ----
    v16h qb[2];
    {
        const float* qp = Q + base + (size_t)(q0 + ln16) * D;
#pragma unroll
        for (int j = 0; j < 2; ++j) {
            const float* p = qp + 32 * j + 16 * hh;
            P16 pk;
#pragma unroll
            for (int c = 0; c < 4; ++c) {
                float4 f = ((const float4*)p)[c];
                pk.p[2*c+0] = __builtin_amdgcn_cvt_pkrtz(f.x * SCALE_LOG2, f.y * SCALE_LOG2);
                pk.p[2*c+1] = __builtin_amdgcn_cvt_pkrtz(f.z * SCALE_LOG2, f.w * SCALE_LOG2);
            }
            qb[j] = pk.v;
        }
    }

    // all-ones B operand: row-sum via WMMA (lands l in the O accumulator layout)
    v16h ones;
#pragma unroll
    for (int e = 0; e < 16; ++e) ones[e] = (_Float16)1.f;

    // accumulators: o[j][nb] holds (q = v + 8*hh, d = ln16 + 16*nb), f32
    v8f o[2][4];
    v8f lacc[2];
#pragma unroll
    for (int j = 0; j < 2; ++j) {
#pragma unroll
        for (int i = 0; i < 8; ++i) lacc[j][i] = 0.f;
#pragma unroll
        for (int nb = 0; nb < 4; ++nb)
#pragma unroll
            for (int i = 0; i < 8; ++i) o[j][nb][i] = 0.f;
    }

    const int skey = tid & 63;          // K staging: one key row / thread
    const int sd0  = (tid >> 6) * 16;   // K staging: 16-d slice
    const int vkq  = tid & 15;          // V staging: 4-key quad
    const int vd0  = (tid >> 4) * 4;    // V staging: 4-d slice

    for (int kb = 0; kb < S; kb += 64) {
        __syncthreads();
        // ---------- stage K (row-major f16) + V (transposed f16) ----------
        {
            const float* kp = K + base + (size_t)(kb + skey) * D + sd0;
            float4 g0 = ((const float4*)kp)[0];
            float4 g1 = ((const float4*)kp)[1];
            float4 g2 = ((const float4*)kp)[2];
            float4 g3 = ((const float4*)kp)[3];
            _Float16* krow = &Kl[skey * KSTR + sd0];
            P8 ka, kc;
            ka.p[0] = __builtin_amdgcn_cvt_pkrtz(g0.x, g0.y);
            ka.p[1] = __builtin_amdgcn_cvt_pkrtz(g0.z, g0.w);
            ka.p[2] = __builtin_amdgcn_cvt_pkrtz(g1.x, g1.y);
            ka.p[3] = __builtin_amdgcn_cvt_pkrtz(g1.z, g1.w);
            kc.p[0] = __builtin_amdgcn_cvt_pkrtz(g2.x, g2.y);
            kc.p[1] = __builtin_amdgcn_cvt_pkrtz(g2.z, g2.w);
            kc.p[2] = __builtin_amdgcn_cvt_pkrtz(g3.x, g3.y);
            kc.p[3] = __builtin_amdgcn_cvt_pkrtz(g3.z, g3.w);
            *(v8h*)(krow)     = ka.v;
            *(v8h*)(krow + 8) = kc.v;

            const float* vp = V + base + (size_t)(kb + 4 * vkq) * D + vd0;
            float4 f0 = ((const float4*)(vp        ))[0];
            float4 f1 = ((const float4*)(vp +     D))[0];
            float4 f2 = ((const float4*)(vp + 2 * D))[0];
            float4 f3 = ((const float4*)(vp + 3 * D))[0];
            P4 t;
            t.p[0] = __builtin_amdgcn_cvt_pkrtz(f0.x, f1.x);
            t.p[1] = __builtin_amdgcn_cvt_pkrtz(f2.x, f3.x);
            *(v4h*)(&Vt[(vd0 + 0) * VSTR + 4 * vkq]) = t.v;
            t.p[0] = __builtin_amdgcn_cvt_pkrtz(f0.y, f1.y);
            t.p[1] = __builtin_amdgcn_cvt_pkrtz(f2.y, f3.y);
            *(v4h*)(&Vt[(vd0 + 1) * VSTR + 4 * vkq]) = t.v;
            t.p[0] = __builtin_amdgcn_cvt_pkrtz(f0.z, f1.z);
            t.p[1] = __builtin_amdgcn_cvt_pkrtz(f2.z, f3.z);
            *(v4h*)(&Vt[(vd0 + 2) * VSTR + 4 * vkq]) = t.v;
            t.p[0] = __builtin_amdgcn_cvt_pkrtz(f0.w, f1.w);
            t.p[1] = __builtin_amdgcn_cvt_pkrtz(f2.w, f3.w);
            *(v4h*)(&Vt[(vd0 + 3) * VSTR + 4 * vkq]) = t.v;

            if (kb + 64 < S) {                       // prefetch next tile
                __builtin_prefetch(kp + 64 * D, 0, 1);
                __builtin_prefetch(vp + 64 * D, 0, 1);
            }
        }
        __syncthreads();

        // ---------------- two 32-key chunks per tile ----------------
#pragma unroll
        for (int ch = 0; ch < 2; ++ch) {
            const int cb = 32 * ch;
            v16h ap[2];
#pragma unroll
            for (int j = 0; j < 2; ++j) {
                // A operand = K rows (16 keys x 32 d): contiguous 16 f16 at
                // offset 32j+16*hh within each key row.
                const _Float16* r0 = &Kl[(cb + ln16) * KSTR + 32 * j + 16 * hh];
                U16 a0, a1;
                a0.h[0] = *(const v8h*)(r0);
                a0.h[1] = *(const v8h*)(r0 + 8);
                a1.h[0] = *(const v8h*)(r0 + 16 * KSTR);
                a1.h[1] = *(const v8h*)(r0 + 16 * KSTR + 8);
                v8f z;
#pragma unroll
                for (int i = 0; i < 8; ++i) z[i] = 0.f;
                // S^T tiles: D holds q = ln16, key = cb + v + 8*hh (+16 for s1)
                v8f s0 = __builtin_amdgcn_wmma_f32_16x16x32_f16(
                             false, a0.v, false, qb[j], (short)0, z, false, false);
                v8f s1 = __builtin_amdgcn_wmma_f32_16x16x32_f16(
                             false, a1.v, false, qb[j], (short)0, z, false, false);
                // fixed-base softmax numerator: raw v_exp_f32 (scores are far
                // from the +/-126 flush range, no libm fixup needed)
                P16 pk;
#pragma unroll
                for (int v = 0; v < 4; ++v) {
                    pk.p[v]     = __builtin_amdgcn_cvt_pkrtz(
                                      __builtin_amdgcn_exp2f(s0[2*v]),
                                      __builtin_amdgcn_exp2f(s0[2*v+1]));
                    pk.p[4 + v] = __builtin_amdgcn_cvt_pkrtz(
                                      __builtin_amdgcn_exp2f(s1[2*v]),
                                      __builtin_amdgcn_exp2f(s1[2*v+1]));
                }
                ap[j] = pk.v;
                // row sums l += P x ones  (same layout as O accumulators)
                lacc[j] = __builtin_amdgcn_wmma_f32_16x16x32_f16(
                              false, ap[j], false, ones, (short)0, lacc[j], false, false);
            }
            // P x V: B operand = Vt rows (8 f16 at key cb + 8*hh, 8 at cb + 16 + 8*hh)
#pragma unroll
            for (int nb = 0; nb < 4; ++nb) {
                const _Float16* vr = &Vt[(16 * nb + ln16) * VSTR + cb + 8 * hh];
                U16 vb;
                vb.h[0] = *(const v8h*)(vr);
                vb.h[1] = *(const v8h*)(vr + 16);
                o[0][nb] = __builtin_amdgcn_wmma_f32_16x16x32_f16(
                               false, ap[0], false, vb.v, (short)0, o[0][nb], false, false);
                o[1][nb] = __builtin_amdgcn_wmma_f32_16x16x32_f16(
                               false, ap[1], false, vb.v, (short)0, o[1][nb], false, false);
            }
        }
    }

    // ---------------- epilogue: combine, layer-norm, store ----------------
    // lacc already sits in the O layout: lacc[j][v] = row-sum for q = v + 8*hh
    float r[4][8];
#pragma unroll
    for (int v = 0; v < 8; ++v) {
        float i0 = __builtin_amdgcn_rcpf(lacc[0][v]);
        float i1 = __builtin_amdgcn_rcpf(lacc[1][v]);
#pragma unroll
        for (int nb = 0; nb < 4; ++nb)
            r[nb][v] = o[0][nb][v] * i0 - lam * (o[1][nb][v] * i1);
    }

#pragma unroll
    for (int v = 0; v < 8; ++v) {
        float sm = 0.f, sq = 0.f;
#pragma unroll
        for (int nb = 0; nb < 4; ++nb) { sm += r[nb][v]; sq += r[nb][v] * r[nb][v]; }
#pragma unroll
        for (int mk = 1; mk < 16; mk <<= 1) {   // reduce over D across 16 lanes (same half)
            sm += __shfl_xor(sm, mk, 32);
            sq += __shfl_xor(sq, mk, 32);
        }
        float mean = sm * (1.f / 64.f);
        float var  = sq * (1.f / 64.f) - mean * mean;
        float rsb  = __builtin_amdgcn_rsqf(var + 1e-5f) * 0.2f;  // *(1 - LAMBDA_INIT)
        float* op = Out + base + (size_t)(q0 + v + 8 * hh) * D + ln16;
#pragma unroll
        for (int nb = 0; nb < 4; ++nb)
            op[16 * nb] = (r[nb][v] - mean) * rsb;
    }
}

extern "C" void kernel_launch(void* const* d_in, const int* in_sizes, int n_in,
                              void* d_out, int out_size, void* d_ws, size_t ws_size,
                              hipStream_t stream) {
    const float* Q   = (const float*)d_in[0];
    const float* K   = (const float*)d_in[1];
    const float* V   = (const float*)d_in[2];
    const float* lam = (const float*)d_in[3];
    float* out       = (float*)d_out;
    dim3 grid(2048 / 128, 2 * 16);   // 16 q-blocks x 32 (b,h)
    dim3 block(256);                 // 8 wave32s
    diffattn_wmma<<<grid, block, 0, stream>>>(Q, K, V, lam, out);
}
